// DelayedMLP_24653112279040
// MI455X (gfx1250) — matrix operand order repaired and checked by reading, not compiled, not run on hardware
//
#include <hip/hip_runtime.h>

// ---------------------------------------------------------------------------
// Problem dims (compile-time constants from the reference)
// ---------------------------------------------------------------------------
#define BATCH 8
#define SEQ   2048
#define IDIM  1024
#define HDIM  4096
#define ODIM  1024
#define MROWS (BATCH * SEQ)   // 16384

typedef __attribute__((ext_vector_type(16))) __bf16 v16bf;
typedef __attribute__((ext_vector_type(8)))  __bf16 v8bf;
typedef __attribute__((ext_vector_type(8)))  float  v8f;
typedef __attribute__((ext_vector_type(4)))  int    v4i;

union Frag16 { v16bf v; v8bf h[2]; };

#define AS3 __attribute__((address_space(3)))
#define AS1 __attribute__((address_space(1)))

// --- CDNA5 async global->LDS copy path (guarded; falls back to VGPR staging)
#if defined(__has_builtin)
#if __has_builtin(__builtin_amdgcn_global_load_async_to_lds_b128)
#define USE_ASYNC_LDS 1
#endif
#endif

#if defined(USE_ASYNC_LDS)
// signature (from clang diagnostic): (global v4i*, shared v4i*, imm offset, imm cpol)
#define ASYNC_LOAD_B128(gp, lp) \
    __builtin_amdgcn_global_load_async_to_lds_b128((AS1 v4i*)(gp), (AS3 v4i*)(lp), 0, 0)
#if __has_builtin(__builtin_amdgcn_s_wait_asynccnt)
#define WAIT_ASYNC0() __builtin_amdgcn_s_wait_asynccnt(0)
#else
#define WAIT_ASYNC0() asm volatile("s_wait_asynccnt 0" ::: "memory")
#endif
#endif

// ---------------------------------------------------------------------------
// fp32 -> bf16 conversion (RNE via native cast)
// ---------------------------------------------------------------------------
__global__ __launch_bounds__(256) void f2bf_kernel(const float* __restrict__ src,
                                                   __bf16* __restrict__ dst,
                                                   size_t n) {
    size_t idx = (size_t)blockIdx.x * blockDim.x + threadIdx.x;
    size_t stride = (size_t)gridDim.x * blockDim.x;
    for (size_t i = idx * 4; i < n; i += stride * 4) {
        float4 f = *(const float4*)(src + i);
        dst[i + 0] = (__bf16)f.x;
        dst[i + 1] = (__bf16)f.y;
        dst[i + 2] = (__bf16)f.z;
        dst[i + 3] = (__bf16)f.w;
    }
}

// ---------------------------------------------------------------------------
// Tiled WMMA bf16 GEMM:  Out[M,N] = epilogue( A[M,K] @ W[N,K]^T + bias[N] )
// Block tile 128x128, BK=32, 256 threads = 8 wave32 waves in a 2(M) x 4(N)
// wave grid; each wave owns a 64x32 sub-tile = 4x2 WMMA 16x16 accumulators.
// LDS double-buffered, row pitch 40 bf16 (80B) => conflict-free ds_load_b128.
// Tile fill uses GLOBAL_LOAD_ASYNC_TO_LDS_B128 (ASYNCcnt) when available.
// OP: 0 = sigmoid -> bf16, 1 = relu -> bf16, 2 = identity -> fp32
// ---------------------------------------------------------------------------
#define LDS_PITCH 40   // bf16 elements per row (32 data + 8 pad)

template <int OP>
__global__ __launch_bounds__(256) void gemm_wmma_bf16(
    const __bf16* __restrict__ A,    // [M,K] row-major
    const __bf16* __restrict__ W,    // [N,K] row-major (i.e. B^T layout)
    const float*  __restrict__ bias, // [N]
    void* __restrict__ Out,          // bf16 (OP 0/1) or fp32 (OP 2), [M,N]
    int M, int N, int K)
{
    __shared__ __bf16 ldsA[2][128 * LDS_PITCH];
    __shared__ __bf16 ldsB[2][128 * LDS_PITCH];

    const int t    = threadIdx.x;
    const int lane = t & 31;
    const int wave = t >> 5;
    const int wm   = wave & 1;   // 0..1  (M direction, 64 rows each)
    const int wn   = wave >> 1;  // 0..3  (N direction, 32 cols each)

    const int m0 = blockIdx.x * 128;
    const int n0 = blockIdx.y * 128;

    // Cooperative tile fill: each thread moves four 16B chunks per k-step
    // (2 for the A tile, 2 for the B tile).
    const int lrow  = t >> 2;        // 0..63
    const int lcol8 = (t & 3) * 8;   // 0,8,16,24 (bf16 elements)

    const __bf16* gA0 = A + (size_t)(m0 + lrow)      * K + lcol8;
    const __bf16* gA1 = A + (size_t)(m0 + 64 + lrow) * K + lcol8;
    const __bf16* gB0 = W + (size_t)(n0 + lrow)      * K + lcol8;
    const __bf16* gB1 = W + (size_t)(n0 + 64 + lrow) * K + lcol8;

    v8f acc[4][2];
#pragma unroll
    for (int mt = 0; mt < 4; ++mt)
#pragma unroll
        for (int nt = 0; nt < 2; ++nt)
            acc[mt][nt] = (v8f){0.f,0.f,0.f,0.f,0.f,0.f,0.f,0.f};

    const int nk = K / 32;

    const int rA  = lane & 15;          // row within 16x16 tile
    const int hi  = lane >> 4;          // half-wave select
    const int koA = hi * 8;             // A frag: K chunks at koA and koA+16
    const int koB = hi * 8;             // B frag: K chunks at koB and koB+16

#if defined(USE_ASYNC_LDS)
    // ---------------- async DMA path: global -> LDS, no VGPR staging -------
    {
        // prologue: tile 0 into buffer 0
        ASYNC_LOAD_B128(gA0, &ldsA[0][lrow * LDS_PITCH + lcol8]);
        ASYNC_LOAD_B128(gA1, &ldsA[0][(lrow + 64) * LDS_PITCH + lcol8]);
        ASYNC_LOAD_B128(gB0, &ldsB[0][lrow * LDS_PITCH + lcol8]);
        ASYNC_LOAD_B128(gB1, &ldsB[0][(lrow + 64) * LDS_PITCH + lcol8]);
    }
    for (int ks = 0; ks < nk; ++ks) {
        const int cur = ks & 1;

        WAIT_ASYNC0();        // our chunks of tile `cur` are in LDS
        __syncthreads();      // everyone's chunks are in LDS; buf cur^1 free

        if (ks + 1 < nk) {
            const int nxt = cur ^ 1;
            const size_t koff = (size_t)(ks + 1) * 32;
            ASYNC_LOAD_B128(gA0 + koff, &ldsA[nxt][lrow * LDS_PITCH + lcol8]);
            ASYNC_LOAD_B128(gA1 + koff, &ldsA[nxt][(lrow + 64) * LDS_PITCH + lcol8]);
            ASYNC_LOAD_B128(gB0 + koff, &ldsB[nxt][lrow * LDS_PITCH + lcol8]);
            ASYNC_LOAD_B128(gB1 + koff, &ldsB[nxt][(lrow + 64) * LDS_PITCH + lcol8]);
        }

        const __bf16* tA = ldsA[cur];
        const __bf16* tB = ldsB[cur];

        v16bf afr[4], bfr[2];
#pragma unroll
        for (int mt = 0; mt < 4; ++mt) {
            const __bf16* p = tA + (wm * 64 + mt * 16 + rA) * LDS_PITCH;
            Frag16 f;
            f.h[0] = *(const v8bf*)(p + koA);
            f.h[1] = *(const v8bf*)(p + koA + 16);
            afr[mt] = f.v;
        }
#pragma unroll
        for (int nt = 0; nt < 2; ++nt) {
            const __bf16* p = tB + (wn * 32 + nt * 16 + rA) * LDS_PITCH + koB;
            Frag16 f;
            f.h[0] = *(const v8bf*)(p);
            f.h[1] = *(const v8bf*)(p + 16);
            bfr[nt] = f.v;
        }

#pragma unroll
        for (int mt = 0; mt < 4; ++mt)
#pragma unroll
            for (int nt = 0; nt < 2; ++nt)
                acc[mt][nt] = __builtin_amdgcn_wmma_f32_16x16x32_bf16(
                    false, afr[mt], false, bfr[nt],
                    (short)0, acc[mt][nt], false, false);
    }
#else
    // ---------------- fallback: VGPR staging path --------------------------
    uint4 ra0, ra1, rb0, rb1;
    ra0 = *(const uint4*)(gA0);
    ra1 = *(const uint4*)(gA1);
    rb0 = *(const uint4*)(gB0);
    rb1 = *(const uint4*)(gB1);
    *(uint4*)(&ldsA[0][lrow * LDS_PITCH + lcol8])        = ra0;
    *(uint4*)(&ldsA[0][(lrow + 64) * LDS_PITCH + lcol8]) = ra1;
    *(uint4*)(&ldsB[0][lrow * LDS_PITCH + lcol8])        = rb0;
    *(uint4*)(&ldsB[0][(lrow + 64) * LDS_PITCH + lcol8]) = rb1;

    for (int ks = 0; ks < nk; ++ks) {
        const int cur = ks & 1;

        if (ks + 1 < nk) {
            const size_t koff = (size_t)(ks + 1) * 32;
            ra0 = *(const uint4*)(gA0 + koff);
            ra1 = *(const uint4*)(gA1 + koff);
            rb0 = *(const uint4*)(gB0 + koff);
            rb1 = *(const uint4*)(gB1 + koff);
        }

        __syncthreads();

        const __bf16* tA = ldsA[cur];
        const __bf16* tB = ldsB[cur];

        v16bf afr[4], bfr[2];
#pragma unroll
        for (int mt = 0; mt < 4; ++mt) {
            const __bf16* p = tA + (wm * 64 + mt * 16 + rA) * LDS_PITCH;
            Frag16 f;
            f.h[0] = *(const v8bf*)(p + koA);
            f.h[1] = *(const v8bf*)(p + koA + 16);
            afr[mt] = f.v;
        }
#pragma unroll
        for (int nt = 0; nt < 2; ++nt) {
            const __bf16* p = tB + (wn * 32 + nt * 16 + rA) * LDS_PITCH + koB;
            Frag16 f;
            f.h[0] = *(const v8bf*)(p);
            f.h[1] = *(const v8bf*)(p + 16);
            bfr[nt] = f.v;
        }

#pragma unroll
        for (int mt = 0; mt < 4; ++mt)
#pragma unroll
            for (int nt = 0; nt < 2; ++nt)
                acc[mt][nt] = __builtin_amdgcn_wmma_f32_16x16x32_bf16(
                    false, afr[mt], false, bfr[nt],
                    (short)0, acc[mt][nt], false, false);

        if (ks + 1 < nk) {
            const int nxt = cur ^ 1;
            *(uint4*)(&ldsA[nxt][lrow * LDS_PITCH + lcol8])        = ra0;
            *(uint4*)(&ldsA[nxt][(lrow + 64) * LDS_PITCH + lcol8]) = ra1;
            *(uint4*)(&ldsB[nxt][lrow * LDS_PITCH + lcol8])        = rb0;
            *(uint4*)(&ldsB[nxt][(lrow + 64) * LDS_PITCH + lcol8]) = rb1;
        }
    }
#endif

    // Epilogue. C/D layout: VGPR r, lanes 0-15 -> M = r, lanes 16-31 -> M = r+8.
#pragma unroll
    for (int mt = 0; mt < 4; ++mt) {
#pragma unroll
        for (int nt = 0; nt < 2; ++nt) {
            const int col   = n0 + wn * 32 + nt * 16 + rA;
            const float bv  = bias[col];
            const int rbase = m0 + wm * 64 + mt * 16 + hi * 8;
#pragma unroll
            for (int r = 0; r < 8; ++r) {
                float v = acc[mt][nt][r] + bv;
                const size_t o = (size_t)(rbase + r) * N + col;
                if constexpr (OP == 0) {
                    float d = 1.0f / (1.0f + __expf(-v));
                    ((__bf16*)Out)[o] = (__bf16)d;
                } else if constexpr (OP == 1) {
                    ((__bf16*)Out)[o] = (__bf16)fmaxf(v, 0.0f);
                } else {
                    ((float*)Out)[o] = v;
                }
            }
        }
    }
}

// ---------------------------------------------------------------------------
// Sequential recurrence, fused elementwise:
//   buf_t = buf_{t-1} * d_t + (1-d_t)*x_t ;  combined_t = x_t*d_t + buf_t
// One thread per (b, i); accesses coalesced across i. Writes bf16 for MLP1.
// ---------------------------------------------------------------------------
__global__ __launch_bounds__(256) void scan_kernel(
    const float* __restrict__ x,       // [B,S,I] fp32
    const __bf16* __restrict__ decay,  // [B,S,I] bf16
    __bf16* __restrict__ comb)         // [B,S,I] bf16
{
    const int i = blockIdx.x * blockDim.x + threadIdx.x;  // 0..IDIM-1
    const int b = blockIdx.y;
    size_t idx = (size_t)b * SEQ * IDIM + i;
    float buf = 0.0f;
    for (int s = 0; s < SEQ; ++s, idx += IDIM) {
        const float d  = (float)decay[idx];
        const float xv = x[idx];
        buf = buf * d + (1.0f - d) * xv;
        comb[idx] = (__bf16)(xv * d + buf);
    }
}

// ---------------------------------------------------------------------------
// Launch
// ---------------------------------------------------------------------------
extern "C" void kernel_launch(void* const* d_in, const int* in_sizes, int n_in,
                              void* d_out, int out_size, void* d_ws, size_t ws_size,
                              hipStream_t stream) {
    const float* x  = (const float*)d_in[0];  // [B,S,I]
    const float* W1 = (const float*)d_in[1];  // [H,I]
    const float* b1 = (const float*)d_in[2];  // [H]
    const float* W2 = (const float*)d_in[3];  // [O,H]
    const float* b2 = (const float*)d_in[4];  // [O]
    const float* Wg = (const float*)d_in[5];  // [I,I]
    const float* bg = (const float*)d_in[6];  // [I]

    char* ws = (char*)d_ws;
    size_t off = 0;
    __bf16* xbf   = (__bf16*)(ws + off); off += (size_t)MROWS * IDIM * 2; // 32 MiB
    __bf16* Wgbf  = (__bf16*)(ws + off); off += (size_t)IDIM * IDIM * 2;  //  2 MiB
    __bf16* W1bf  = (__bf16*)(ws + off); off += (size_t)HDIM * IDIM * 2;  //  8 MiB
    __bf16* W2bf  = (__bf16*)(ws + off); off += (size_t)ODIM * HDIM * 2;  //  8 MiB
    __bf16* decay = (__bf16*)(ws + off); off += (size_t)MROWS * IDIM * 2; // 32 MiB
    __bf16* comb  = (__bf16*)(ws + off); off += (size_t)MROWS * IDIM * 2; // 32 MiB
    __bf16* hid   = (__bf16*)(ws + off); off += (size_t)MROWS * HDIM * 2; // 128 MiB

    // 1) convert activations + weights to bf16
    f2bf_kernel<<<4096, 256, 0, stream>>>(x,  xbf,  (size_t)MROWS * IDIM);
    f2bf_kernel<<<1024, 256, 0, stream>>>(Wg, Wgbf, (size_t)IDIM * IDIM);
    f2bf_kernel<<<2048, 256, 0, stream>>>(W1, W1bf, (size_t)HDIM * IDIM);
    f2bf_kernel<<<2048, 256, 0, stream>>>(W2, W2bf, (size_t)ODIM * HDIM);

    // 2) gate GEMM + sigmoid: decay = sigmoid(x @ Wg^T + bg)
    gemm_wmma_bf16<0><<<dim3(MROWS / 128, IDIM / 128), 256, 0, stream>>>(
        xbf, Wgbf, bg, decay, MROWS, IDIM, IDIM);

    // 3) fused recurrence -> combined (bf16)
    scan_kernel<<<dim3(IDIM / 256, BATCH), 256, 0, stream>>>(x, decay, comb);

    // 4) hidden = relu(combined @ W1^T + b1)  (bf16 out)
    gemm_wmma_bf16<1><<<dim3(MROWS / 128, HDIM / 128), 256, 0, stream>>>(
        comb, W1bf, b1, hid, MROWS, HDIM, IDIM);

    // 5) out = hidden @ W2^T + b2  (fp32 out)
    gemm_wmma_bf16<2><<<dim3(MROWS / 128, ODIM / 128), 256, 0, stream>>>(
        hid, W2bf, b2, d_out, MROWS, ODIM, HDIM);
}
